// ESAMoleculeClassifier_6691559047219
// MI455X (gfx1250) — compile-verified, matrix operand order checked by reading, not run on hardware
//
#include <hip/hip_runtime.h>
#include <hip/hip_bf16.h>

// ESA molecule classifier for MI455X (gfx1250, wave32).
// All matmuls on V_WMMA_F32_16X16X32_F16 (f16 operands, f32 accum):
//  - GEMMs: 64x64 block tile, 4 waves, 2x2 WMMA per wave per K-step,
//    LDS double-buffered; A tile staged with GLOBAL_LOAD_ASYNC_TO_LDS_B128
//    (ASYNCcnt pipeline) when the toolchain exposes it; B staged transposed.
//  - Attention: dh=32 == WMMA K-depth -> one WMMA per 16x16 score tile;
//    V projection written pre-transposed so P@V fragments are contiguous.
// Softmax / LayerNorm / residual adds in fp32.

typedef float v8f __attribute__((ext_vector_type(8)));
typedef _Float16 half_t;
typedef _Float16 v16h __attribute__((ext_vector_type(16)));
typedef int v4i __attribute__((ext_vector_type(4)));

union FragU { uint4 u[2]; v16h h; };

#define HID 256
#define NHEAD 8
#define DH 32
#define NB 32
#define LK 512

#if defined(__AMDGCN__) && __has_builtin(__builtin_amdgcn_global_load_async_to_lds_b128)
#define ASYNC_COPY 1
#else
#define ASYNC_COPY 0
#endif

#define AS_GLOBAL __attribute__((address_space(1)))
#define AS_LOCAL  __attribute__((address_space(3)))

static __device__ __forceinline__ void async_copy16B(const half_t* g, half_t* l)
{
#if ASYNC_COPY
    __builtin_amdgcn_global_load_async_to_lds_b128(
        (AS_GLOBAL v4i*)g, (AS_LOCAL v4i*)l, 0, 0);
#else
    *(uint4*)l = *(const uint4*)g;
#endif
}

static __device__ __forceinline__ void async_wait0()
{
#if ASYNC_COPY
#if __has_builtin(__builtin_amdgcn_s_wait_asynccnt)
    __builtin_amdgcn_s_wait_asynccnt(0);
#else
    asm volatile("s_wait_asynccnt 0x0" ::: "memory");
#endif
#endif
}

// ---------------------------------------------------------------------------
// f32 -> f16 elementwise convert (inputs / weights)
// ---------------------------------------------------------------------------
__global__ __launch_bounds__(256)
void cvt16_k(const float* __restrict__ in, half_t* __restrict__ out, int n)
{
    int i = blockIdx.x * 256 + threadIdx.x;
    if (i < n) out[i] = (half_t)in[i];
}

// ---------------------------------------------------------------------------
// f16 GEMM: out = act(A[row % row_mod, K] @ W[K,N] + bias [+ res])
// Block 128 thr (4 waves); block tile 64x64; wave computes 2x2 16x16 tiles.
// K-step 32, LDS double buffered. vtrans: write f16 out as [b][h][d][key].
// Requires N%64==0, K%8==0.
// ---------------------------------------------------------------------------
__global__ __launch_bounds__(128)
void linear16_k(const half_t* __restrict__ A, const half_t* __restrict__ W,
                const float* __restrict__ bias,
                const float* __restrict__ resf, const half_t* __restrict__ resh,
                float* __restrict__ outf, half_t* __restrict__ outh,
                int M, int N, int K, int row_mod, int relu, int vtrans)
{
    __shared__ __align__(16) half_t sA[2][64][32];   // [buf][row][k]
    __shared__ __align__(16) half_t sB[2][64][32];   // [buf][n][k] (transposed)

    const int tid   = threadIdx.x;
    const int wave  = tid >> 5;
    const int lane  = tid & 31;
    const int row16 = lane & 15;
    const int hs    = lane >> 4;
    const int row0  = blockIdx.y * 64;
    const int col0  = blockIdx.x * 64;
    const int mloc  = (wave >> 1) * 32;
    const int nloc  = (wave & 1) * 32;

    const int sar = tid >> 1, sah = tid & 1;   // A staging: row, 16-half half
    const int sbk = tid >> 2, sbq = tid & 3;   // B staging: k-row, 16-col quarter
    const int aRowG = (row0 + sar) % row_mod;
    const int KT = (K + 31) >> 5;
    const uint4 z4 = {0u, 0u, 0u, 0u};

    auto stage = [&](int b, int kt) {
        const int kk = kt << 5;
        {   // A tile: row sar, halves kk+sah*16 .. +15 (K always multiple of 8)
            const int k = kk + sah * 16;
            const half_t* arp = A + (size_t)aRowG * K + k;
            half_t* dst = &sA[b][sar][sah * 16];
            if (k + 16 <= K) {                 // fully in range: async DMA to LDS
                async_copy16B(arp,     dst);
                async_copy16B(arp + 8, dst + 8);
            } else if (k + 8 <= K) {           // half in range
                async_copy16B(arp, dst);
                *(uint4*)(dst + 8) = z4;
            } else {                            // fully out of range (K < 32 tail)
                *(uint4*)(dst)     = z4;
                *(uint4*)(dst + 8) = z4;
            }
        }
        {   // B tile: W row kk+sbk, 16 cols -> transpose scatter
            const int k = kk + sbk;
            if (k < K) {
                const half_t* wr = W + (size_t)k * N + col0 + sbq * 16;
#pragma unroll
                for (int i = 0; i < 16; ++i) sB[b][sbq * 16 + i][sbk] = wr[i];
                if (kk + 32 + sbk < K)
                    __builtin_prefetch(W + (size_t)(kk + 32 + sbk) * N + col0, 0, 0);
            } else {
#pragma unroll
                for (int i = 0; i < 16; ++i) sB[b][sbq * 16 + i][sbk] = (half_t)0.0f;
            }
        }
    };

    v8f acc00 = {}, acc01 = {}, acc10 = {}, acc11 = {};
    int buf = 0;
    stage(0, 0);
    async_wait0();
    __syncthreads();
    for (int kt = 0; kt < KT; ++kt) {
        if (kt + 1 < KT) stage(buf ^ 1, kt + 1);
        const half_t* ar0 = &sA[buf][mloc + row16][0];
        const half_t* ar1 = &sA[buf][mloc + 16 + row16][0];
        const half_t* br0 = &sB[buf][nloc + row16][0];
        const half_t* br1 = &sB[buf][nloc + 16 + row16][0];
        FragU a0, a1, b0, b1;
        a0.u[0] = *(const uint4*)(ar0 + hs * 8); a0.u[1] = *(const uint4*)(ar0 + hs * 8 + 16);
        a1.u[0] = *(const uint4*)(ar1 + hs * 8); a1.u[1] = *(const uint4*)(ar1 + hs * 8 + 16);
        b0.u[0] = *(const uint4*)(br0 + hs * 8); b0.u[1] = *(const uint4*)(br0 + hs * 8 + 16);
        b1.u[0] = *(const uint4*)(br1 + hs * 8); b1.u[1] = *(const uint4*)(br1 + hs * 8 + 16);
        acc00 = __builtin_amdgcn_wmma_f32_16x16x32_f16(false, a0.h, false, b0.h, (short)0, acc00, false, false);
        acc01 = __builtin_amdgcn_wmma_f32_16x16x32_f16(false, a0.h, false, b1.h, (short)0, acc01, false, false);
        acc10 = __builtin_amdgcn_wmma_f32_16x16x32_f16(false, a1.h, false, b0.h, (short)0, acc10, false, false);
        acc11 = __builtin_amdgcn_wmma_f32_16x16x32_f16(false, a1.h, false, b1.h, (short)0, acc11, false, false);
        async_wait0();
        __syncthreads();
        buf ^= 1;
    }

    auto emit = [&](const v8f& acc, int msub, int nsub) {
        const int gc = col0 + nloc + nsub * 16 + row16;
        const float bv = bias ? bias[gc] : 0.0f;
#pragma unroll
        for (int r = 0; r < 8; ++r) {
            const int gr = row0 + mloc + msub * 16 + hs * 8 + r;
            if (gr < M) {
                float v = acc[r] + bv;
                if (resf) v += resf[(size_t)gr * N + gc];
                if (resh) v += (float)resh[(size_t)gr * N + gc];
                if (relu) v = fmaxf(v, 0.0f);
                if (outf) outf[(size_t)gr * N + gc] = v;
                if (outh) {
                    if (vtrans) {   // [b][head][d][key], key-major rows of LK
                        const int bb = gr >> 9, key = gr & (LK - 1);
                        const int hdh = gc >> 5, dd = gc & (DH - 1);
                        outh[(((size_t)bb * NHEAD + hdh) * DH + dd) * LK + key] = (half_t)v;
                    } else {
                        outh[(size_t)gr * N + gc] = (half_t)v;
                    }
                }
            }
        }
    };
    emit(acc00, 0, 0); emit(acc01, 0, 1); emit(acc10, 1, 0); emit(acc11, 1, 1);
}

// ---------------------------------------------------------------------------
// cat16[b,i,:] = [ h16[b,src], h16[b,dst], e16[b,i] ]
// ---------------------------------------------------------------------------
__global__ __launch_bounds__(256)
void gather_k(const half_t* __restrict__ h, const half_t* __restrict__ e,
              const int* __restrict__ ei, half_t* __restrict__ cat)
{
    const int blk = blockIdx.x;
    const int b   = blk >> 9;
    const int i   = blk & 511;
    const int t   = threadIdx.x;
    const int src = ei[b * 1024 + i];
    const int dst = ei[b * 1024 + 512 + i];
    const size_t crow = (size_t)blk * 768;
    cat[crow + t]       = h[((size_t)b * 128 + src) * HID + t];
    cat[crow + 256 + t] = h[((size_t)b * 128 + dst) * HID + t];
    cat[crow + 512 + t] = e[(size_t)blk * HID + t];
}

// ---------------------------------------------------------------------------
// Attention, f16 WMMA.  One wave per (16-query block, head, batch).
// Q,K: [b][row][256] f16.  Vt: [b][head][d][key] f16 (pre-transposed).
// dh == 32 == WMMA K-depth: one WMMA per 16x16 score tile.
// ---------------------------------------------------------------------------
__global__ __launch_bounds__(32)
void attn16_k(const half_t* __restrict__ Q, const half_t* __restrict__ Kp,
              const half_t* __restrict__ Vt, float* __restrict__ O, int Lq)
{
    __shared__ float  s[16][LK];                 // 32 KB fp32 scores
    __shared__ __align__(16) half_t p[16][LK];   // 16 KB f16 probabilities

    const int lane  = threadIdx.x;
    const int row16 = lane & 15;
    const int hs    = lane >> 4;
    const int hi8   = hs << 3;
    const int q0 = blockIdx.x * 16;
    const int hh = blockIdx.y;
    const int b  = blockIdx.z;

    int qrow = q0 + row16;
    if (qrow >= Lq) qrow = Lq - 1;               // PMA clamp; stores guarded
    const half_t* qp = Q + ((size_t)b * Lq + qrow) * HID + hh * DH;
    FragU qa;
    qa.u[0] = *(const uint4*)(qp + hs * 8);
    qa.u[1] = *(const uint4*)(qp + hs * 8 + 16);

    // ---- scores: 32 key tiles, ONE f16 WMMA each (K = dh = 32) ----
    for (int j = 0; j < 32; ++j) {
        const half_t* kp = Kp + ((size_t)b * LK + j * 16 + row16) * HID + hh * DH;
        FragU kb;
        kb.u[0] = *(const uint4*)(kp + hs * 8);
        kb.u[1] = *(const uint4*)(kp + hs * 8 + 16);
        v8f acc = {};
        acc = __builtin_amdgcn_wmma_f32_16x16x32_f16(
            false, qa.h, false, kb.h, (short)0, acc, false, false);
#pragma unroll
        for (int r = 0; r < 8; ++r)
            s[r + hi8][j * 16 + row16] = acc[r] * 0.17677669529663687f; // 1/sqrt(32)
    }
    __syncthreads();

    // ---- fp32 softmax over 512 keys; emit f16 probabilities ----
    for (int r = 0; r < 16; ++r) {
        float mx = -3.402823466e38f;
        for (int c = lane; c < LK; c += 32) mx = fmaxf(mx, s[r][c]);
        for (int o = 16; o; o >>= 1) mx = fmaxf(mx, __shfl_xor(mx, o, 32));
        float sum = 0.0f;
        for (int c = lane; c < LK; c += 32) {
            float t = __expf(s[r][c] - mx);
            s[r][c] = t;
            sum += t;
        }
        for (int o = 16; o; o >>= 1) sum += __shfl_xor(sum, o, 32);
        const float inv = 1.0f / sum;
        for (int c = lane; c < LK; c += 32) p[r][c] = (half_t)(s[r][c] * inv);
    }
    __syncthreads();

    // ---- O = P(16x512) @ V(512x32): 16 K-steps x 2 tiles, all contiguous ----
#pragma unroll
    for (int ct = 0; ct < 2; ++ct) {
        const int d = ct * 16 + row16;
        const half_t* vrow = Vt + (((size_t)b * NHEAD + hh) * DH + d) * LK;
        v8f acc = {};
        for (int k0 = 0; k0 < LK; k0 += 32) {
            FragU pa, vb;
            pa.u[0] = *(const uint4*)&p[row16][k0 + hs * 8];
            pa.u[1] = *(const uint4*)&p[row16][k0 + hs * 8 + 16];
            vb.u[0] = *(const uint4*)(vrow + k0 + hs * 8);
            vb.u[1] = *(const uint4*)(vrow + k0 + hs * 8 + 16);
            acc = __builtin_amdgcn_wmma_f32_16x16x32_f16(
                false, pa.h, false, vb.h, (short)0, acc, false, false);
        }
#pragma unroll
        for (int r = 0; r < 8; ++r) {
            const int orow = q0 + r + hi8;
            if (orow < Lq)
                O[((size_t)b * Lq + orow) * HID + hh * DH + ct * 16 + row16] = acc[r];
        }
    }
}

// ---------------------------------------------------------------------------
// out16[row,:] = LayerNorm(x[row,:] (+ res[row,:])) * g + be
// ---------------------------------------------------------------------------
__global__ __launch_bounds__(128)
void add_ln_k(const float* __restrict__ x, const float* __restrict__ res,
              const float* __restrict__ g, const float* __restrict__ be,
              half_t* __restrict__ out, int rows)
{
    const int lane = threadIdx.x & 31;
    const int row  = blockIdx.x * 4 + (threadIdx.x >> 5);
    if (row >= rows) return;
    const size_t base = (size_t)row * HID;
    float v[8];
    float sum = 0.0f, sq = 0.0f;
#pragma unroll
    for (int i = 0; i < 8; ++i) {
        float t = x[base + lane + i * 32];
        if (res) t += res[base + lane + i * 32];
        v[i] = t;
        sum += t;
        sq  += t * t;
    }
    for (int o = 16; o; o >>= 1) {
        sum += __shfl_xor(sum, o, 32);
        sq  += __shfl_xor(sq,  o, 32);
    }
    const float m   = sum * (1.0f / HID);
    const float var = sq * (1.0f / HID) - m * m;
    const float inv = rsqrtf(var + 1e-5f);
#pragma unroll
    for (int i = 0; i < 8; ++i) {
        const int c = lane + i * 32;
        out[base + c] = (half_t)((v[i] - m) * inv * g[c] + be[c]);
    }
}

// ---------------------------------------------------------------------------
// logits[b] = hid16[b,:] . w2 + b2
// ---------------------------------------------------------------------------
__global__ __launch_bounds__(32)
void cls_head_k(const half_t* __restrict__ hid, const float* __restrict__ w,
                const float* __restrict__ b, float* __restrict__ outp)
{
    const int bi = blockIdx.x, lane = threadIdx.x;
    float s = 0.0f;
    for (int i = lane; i < HID; i += 32)
        s += (float)hid[(size_t)bi * HID + i] * w[i];
    for (int o = 16; o; o >>= 1) s += __shfl_xor(s, o, 32);
    if (lane == 0) outp[bi] = s + b[0];
}

// ===========================================================================
// Host-side orchestration
// ===========================================================================
static inline void cvt(const float* in, half_t* out, int n, hipStream_t s)
{
    cvt16_k<<<(n + 255) / 256, 256, 0, s>>>(in, out, n);
}

static inline void lin(const half_t* A, const half_t* W, const float* b,
                       const float* resf, const half_t* resh,
                       float* outf, half_t* outh,
                       int M, int N, int K, int row_mod, int relu, int vtrans,
                       hipStream_t s)
{
    dim3 grid(N / 64, (M + 63) / 64);
    linear16_k<<<grid, 128, 0, s>>>(A, W, b, resf, resh, outf, outh,
                                    M, N, K, row_mod, relu, vtrans);
}

struct MabP {
    const float *qw, *qb, *kw, *kb, *vw, *vb;
    const float *g0, *be0, *g1, *be1;
    const float *r1w, *r1b, *r2w, *r2b;
};

static MabP get_mab(void* const* in, int base)
{
    auto f = [&](int i) { return (const float*)in[base + i]; };
    MabP m;
    m.qw = f(0);  m.qb = f(1);  m.kw = f(2);  m.kb = f(3);
    m.vw = f(4);  m.vb = f(5);  m.g0 = f(6);  m.be0 = f(7);
    m.g1 = f(8);  m.be1 = f(9); m.r1w = f(10); m.r1b = f(11);
    m.r2w = f(12); m.r2b = f(13);
    return m;
}

struct Ws {
    float  *F0, *F1;                               // f32 (16MB each)
    half_t *HE, *HX, *HH, *HQ, *HK, *HV, *HCAT, *HW, *HS;
};

static void run_mab(const MabP& m, const half_t* Qin, int Lq, int row_mod,
                    half_t* Eout, const Ws& w, hipStream_t s)
{
    const int Mq = NB * Lq;
    const int Mk = NB * LK;
    cvt(m.qw, w.HW, HID * HID, s);
    lin(Qin,  w.HW, m.qb, nullptr, nullptr, w.F0, w.HQ, Mq, HID, HID, row_mod, 0, 0, s);
    cvt(m.kw, w.HW, HID * HID, s);
    lin(w.HE, w.HW, m.kb, nullptr, nullptr, nullptr, w.HK, Mk, HID, HID, Mk, 0, 0, s);
    cvt(m.vw, w.HW, HID * HID, s);
    lin(w.HE, w.HW, m.vb, nullptr, nullptr, nullptr, w.HV, Mk, HID, HID, Mk, 0, 1, s);
    attn16_k<<<dim3((Lq + 15) / 16, NHEAD, NB), 32, 0, s>>>(w.HQ, w.HK, w.HV, w.F1, Lq);
    // X = LN(Qp + o) -> f16
    add_ln_k<<<(Mq + 3) / 4, 128, 0, s>>>(w.F1, w.F0, m.g0, m.be0, w.HX, Mq);
    // X + MLP(X)
    cvt(m.r1w, w.HW, HID * HID, s);
    lin(w.HX, w.HW, m.r1b, nullptr, nullptr, nullptr, w.HH, Mq, HID, HID, Mq, 1, 0, s);
    cvt(m.r2w, w.HW, HID * HID, s);
    lin(w.HH, w.HW, m.r2b, nullptr, w.HX, w.F0, nullptr, Mq, HID, HID, Mq, 0, 0, s);
    add_ln_k<<<(Mq + 3) / 4, 128, 0, s>>>(w.F0, nullptr, m.g1, m.be1, Eout, Mq);
}

extern "C" void kernel_launch(void* const* d_in, const int* in_sizes, int n_in,
                              void* d_out, int out_size, void* d_ws, size_t ws_size,
                              hipStream_t stream)
{
    (void)in_sizes; (void)n_in; (void)out_size; (void)ws_size;

    // Input order (setup_inputs dict insertion order, depth-first):
    // 0:x 1:edge_index 2:edge_attr | 3..6 node | 7..10 edge | 11..14 eset
    // 15..28 sab0 | 29..42 sab1 | 43..56 sab2 | 57 S | 58..71 pma | 72..75 cls
    const float* x  = (const float*)d_in[0];
    const int*   ei = (const int*)d_in[1];
    const float* ea = (const float*)d_in[2];
    auto F = [&](int i) { return (const float*)d_in[i]; };

    // Workspace: F0,F1 (f32, 16MB ea) + f16 pool:
    //  HE,HX,HH,HQ,HK,HV (8MB ea) + HCAT (24MB) + HW (1MB) + HS  ~= 105 MB
    const size_t REG = (size_t)4 * 1024 * 1024;
    Ws w;
    w.F0 = (float*)d_ws;
    w.F1 = w.F0 + REG;
    half_t* hb = (half_t*)(w.F1 + REG);
    w.HE   = hb;
    w.HX   = w.HE + REG;
    w.HH   = w.HX + REG;
    w.HQ   = w.HH + REG;
    w.HK   = w.HQ + REG;
    w.HV   = w.HK + REG;
    w.HCAT = w.HV + REG;          // 12M halves
    w.HW   = w.HCAT + 3 * REG;    // 512K halves (weight scratch)
    w.HS   = w.HW + 512 * 1024;   // 1K halves

    // transient f16 inputs at head of HCAT (dead before cat is built)
    half_t* x16  = w.HCAT;
    half_t* ea16 = w.HCAT + 131072;

    cvt(x,     x16,  NB * 128 * 32, stream);
    cvt(ea,    ea16, NB * 512 * 16, stream);
    cvt(F(57), w.HS, 4 * HID,       stream);   // S

    // ---- node MLP: (4096,32)->256(ReLU)->256 ; h16 -> HE ----
    cvt(F(3), w.HW, 32 * HID, stream);
    lin(x16,  w.HW, F(4), nullptr, nullptr, nullptr, w.HH, 4096, HID, 32,  4096, 1, 0, stream);
    cvt(F(5), w.HW, HID * HID, stream);
    lin(w.HH, w.HW, F(6), nullptr, nullptr, nullptr, w.HE, 4096, HID, HID, 4096, 0, 0, stream);

    // ---- edge MLP: (16384,16)->256(ReLU)->256 ; e16 -> HX ----
    cvt(F(7), w.HW, 16 * HID, stream);
    lin(ea16, w.HW, F(8),  nullptr, nullptr, nullptr, w.HH, 16384, HID, 16,  16384, 1, 0, stream);
    cvt(F(9), w.HW, HID * HID, stream);
    lin(w.HH, w.HW, F(10), nullptr, nullptr, nullptr, w.HX, 16384, HID, HID, 16384, 0, 0, stream);

    // ---- gather + concat (f16) ----
    gather_k<<<16384, 256, 0, stream>>>(w.HE, w.HX, ei, w.HCAT);

    // ---- eset MLP: (16384,768)->256(ReLU)->256 ; E16 -> HE ----
    cvt(F(11), w.HW, 768 * HID, stream);
    lin(w.HCAT, w.HW, F(12), nullptr, nullptr, nullptr, w.HH, 16384, HID, 768, 16384, 1, 0, stream);
    cvt(F(13), w.HW, HID * HID, stream);
    lin(w.HH,   w.HW, F(14), nullptr, nullptr, nullptr, w.HE, 16384, HID, HID, 16384, 0, 0, stream);

    // ---- 3 x SAB ----
    for (int i = 0; i < 3; ++i) {
        MabP m = get_mab(d_in, 15 + 14 * i);
        run_mab(m, w.HE, LK, NB * LK, w.HE, w, stream);
    }

    // ---- PMA: queries = S16 broadcast (row_mod=4); pooled16 -> HH ----
    MabP pma = get_mab(d_in, 58);
    run_mab(pma, w.HS, 4, 4, w.HH, w, stream);

    // ---- classifier: (32,1024)->256(ReLU)->1 ----
    cvt(F(72), w.HW, 1024 * HID, stream);
    lin(w.HH, w.HW, F(73), nullptr, nullptr, nullptr, w.HX, NB, HID, 1024, NB, 1, 0, stream);
    cls_head_k<<<NB, 32, 0, stream>>>(w.HX, F(74), F(75), (float*)d_out);
}
